// FlattenIntraCycleMoELayer_15479062135204
// MI455X (gfx1250) — compile-verified, matrix operand order checked by reading, not run on hardware
//
#include <hip/hip_runtime.h>
#include <hip/hip_bf16.h>

// ---------------------------------------------------------------------------
// FlattenIntraCycleMoELayer for MI455X (gfx1250, wave32, WMMA)
//
//   Kernel 1 (gate): logits = DKP @ W_gate + b_gate, top-p(0.25) mask,
//                    softmax, renormalize -> w[b, 8] in workspace.
//   Kernel 2 (mm):   fold experts per batch:  W_eff[b] = sum_e w[b,e]*W_exp[e]
//                    out[b] = f32(bf16(X_b @ W_eff[b] + b_eff[b]))
//                           + (X_b @ (W_gen0+W_gen1) + b_gen0+b_gen1)
//                    via v_wmma_f32_16x16x32_bf16, f32 accumulators.
//   Gate weights are block-uniform -> readfirstlane so expert skips are
//   scalar branches. K tail (900 = 14*64 + 4) is peeled. f32->bf16 staging
//   uses vector converts so the backend can emit packed cvt instructions.
// ---------------------------------------------------------------------------

typedef __bf16 bf16;
typedef __bf16 v4bf  __attribute__((ext_vector_type(4)));
typedef __bf16 v8bf  __attribute__((ext_vector_type(8)));
typedef __bf16 v16bf __attribute__((ext_vector_type(16)));
typedef float  v4f   __attribute__((ext_vector_type(4)));
typedef float  v8f   __attribute__((ext_vector_type(8)));

#define B_SZ   512
#define L_SZ   128
#define K_SZ   900      // 3*F
#define D_SZ   512      // D_MODEL
#define KC     64       // K chunk
#define K_FULL 896      // 14 full chunks
#define K_REM  4        // tail
#define KPAD   72       // LDS pitch (mult of 8 -> 16B rows, bank spread)
#define MT     128
#define NT     64

// round-to-bf16-precision, result kept in f32 (reference's astype(bf16))
__device__ __forceinline__ float bfround(float f) {
    unsigned u = __float_as_uint(f);
    unsigned r = u + 0x7fffu + ((u >> 16) & 1u);
    return __uint_as_float(r & 0xffff0000u);
}
__device__ __forceinline__ float rfl(float x) {   // uniform value -> SGPR
    return __uint_as_float(__builtin_amdgcn_readfirstlane(__float_as_uint(x)));
}
// WMMA 16-bit fragment: two contiguous 16B LDS loads (K{kb..kb+7},{kb+16..kb+23})
__device__ __forceinline__ v16bf ldfrag(const bf16* p) {
    v8bf lo = *(const v8bf*)(p);
    v8bf hi = *(const v8bf*)(p + 16);
    return __builtin_shufflevector(lo, hi, 0, 1, 2, 3, 4, 5, 6, 7,
                                           8, 9, 10, 11, 12, 13, 14, 15);
}

// ---------------------------------------------------------------------------
// Kernel 1: gating (block per batch row; wave e computes logit e)
// ---------------------------------------------------------------------------
__global__ __launch_bounds__(256) void moe_gate_kernel(
    const float* __restrict__ dkp,   // [512, 4096]
    const float* __restrict__ Wg,    // [4096, 8]
    const float* __restrict__ bg,    // [8]
    float* __restrict__ wOut)        // [512, 8]
{
    const int b    = blockIdx.x;
    const int lane = threadIdx.x & 31;
    const int wave = threadIdx.x >> 5;
    __shared__ float sLogit[8];

    const float* row = dkp + (size_t)b * 4096;
    float acc = 0.f;
    for (int d = lane; d < 4096; d += 32)
        acc += row[d] * Wg[d * 8 + wave];
#pragma unroll
    for (int off = 16; off; off >>= 1)
        acc += __shfl_xor(acc, off, 32);
    if (lane == 0) sLogit[wave] = acc + bg[wave];
    __syncthreads();

    if (threadIdx.x == 0) {
        float p[8];
        float mx = -1e30f;
#pragma unroll
        for (int e = 0; e < 8; ++e) mx = fmaxf(mx, sLogit[e]);
        float s = 0.f;
#pragma unroll
        for (int e = 0; e < 8; ++e) { p[e] = __expf(sLogit[e] - mx); s += p[e]; }
#pragma unroll
        for (int e = 0; e < 8; ++e) p[e] /= s;
        int ord[8];
#pragma unroll
        for (int e = 0; e < 8; ++e) ord[e] = e;
        for (int i = 0; i < 8; ++i) {
            int best = i;
            for (int j = i + 1; j < 8; ++j)
                if (p[ord[j]] > p[ord[best]]) best = j;
            int t = ord[i]; ord[i] = ord[best]; ord[best] = t;
        }
        float keepW[8];
#pragma unroll
        for (int e = 0; e < 8; ++e) keepW[e] = 0.f;
        float cum = 0.f, tot = 0.f;
        for (int j = 0; j < 8; ++j) {
            cum += p[ord[j]];
            if (j == 0 || cum < 0.25f) { keepW[ord[j]] = p[ord[j]]; tot += p[ord[j]]; }
        }
        float inv = 1.f / (tot + 1e-9f);
#pragma unroll
        for (int e = 0; e < 8; ++e)
            wOut[b * 8 + e] = keepW[e] * inv;
    }
}

// ---------------------------------------------------------------------------
// Kernel 2: fused expert+general WMMA GEMM
// grid = (D_SZ/NT, B_SZ); block = 256 threads (8 waves, 4M x 2N wave grid)
// ---------------------------------------------------------------------------
__global__ __launch_bounds__(256) void moe_mm_kernel(
    const float* __restrict__ x,     // [512, 128, 900]
    const float* __restrict__ Wexp,  // [8, 900, 512]
    const float* __restrict__ bexp,  // [8, 512]
    const float* __restrict__ Wgen,  // [2, 900, 512]
    const float* __restrict__ bgen,  // [2, 512]
    const float* __restrict__ wGate, // [512, 8]
    float* __restrict__ out)         // [512, 128, 512] f32
{
    const int n0  = blockIdx.x * NT;
    const int b   = blockIdx.y;
    const int tid = threadIdx.x;

    __shared__ bf16 shA[MT * KPAD];  // X tile,   [m][k]
    __shared__ bf16 shE[NT * KPAD];  // W_eff^T,  [n][k]
    __shared__ bf16 shG[NT * KPAD];  // W_gsum^T, [n][k]

    // gate weights are uniform across the block -> force into SGPRs
    float we[8];
#pragma unroll
    for (int e = 0; e < 8; ++e) we[e] = rfl(wGate[b * 8 + e]);

    const int lane = tid & 31;
    const int wave = tid >> 5;
    const int wm = (wave & 3) * 32;
    const int wn = (wave >> 2) * 32;

    v8f accE[4], accG[4];
#pragma unroll
    for (int i = 0; i < 4; ++i)
#pragma unroll
        for (int j = 0; j < 8; ++j) { accE[i][j] = 0.f; accG[i][j] = 0.f; }

    const float* xb = x + (size_t)b * L_SZ * K_SZ;

    // ---- stage W_eff^T / W_gsum^T quad (4 consecutive n), scalar-branch skips
    auto stageW = [&](int kg, int k, int n) {
        v4f sE;
#pragma unroll
        for (int j = 0; j < 4; ++j) sE[j] = 0.f;
        size_t off = (size_t)kg * D_SZ + (n0 + n);
#pragma unroll
        for (int e = 0; e < 8; ++e) {
            if (we[e] != 0.f) {                 // scalar cond -> s_cbranch
                v4f wv = *(const v4f*)(Wexp + (size_t)e * K_SZ * D_SZ + off);
#pragma unroll
                for (int j = 0; j < 4; ++j) sE[j] += we[e] * wv[j];
            }
        }
        v4f g0 = *(const v4f*)(Wgen + off);
        v4f g1 = *(const v4f*)(Wgen + (size_t)K_SZ * D_SZ + off);
        v4bf hE = __builtin_convertvector(sE, v4bf);        // packed cvt
        v4bf hG = __builtin_convertvector(g0 + g1, v4bf);   // packed cvt
#pragma unroll
        for (int j = 0; j < 4; ++j) {
            shE[(n + j) * KPAD + k] = hE[j];
            shG[(n + j) * KPAD + k] = hG[j];
        }
    };

    auto doWmma = [&]() {
#pragma unroll
        for (int ks = 0; ks < KC; ks += 32) {
            const int kb = ks + ((lane & 16) ? 8 : 0);
            const int rl = lane & 15;
            v16bf aF[2], eF[2], gF[2];
#pragma unroll
            for (int i = 0; i < 2; ++i) {
                aF[i] = ldfrag(&shA[(wm + i * 16 + rl) * KPAD + kb]);
                eF[i] = ldfrag(&shE[(wn + i * 16 + rl) * KPAD + kb]);
                gF[i] = ldfrag(&shG[(wn + i * 16 + rl) * KPAD + kb]);
            }
#pragma unroll
            for (int mi = 0; mi < 2; ++mi)
#pragma unroll
                for (int ni = 0; ni < 2; ++ni) {
                    accE[mi * 2 + ni] = __builtin_amdgcn_wmma_f32_16x16x32_bf16(
                        false, aF[mi], false, eF[ni], (short)0,
                        accE[mi * 2 + ni], false, false);
                    accG[mi * 2 + ni] = __builtin_amdgcn_wmma_f32_16x16x32_bf16(
                        false, aF[mi], false, gF[ni], (short)0,
                        accG[mi * 2 + ni], false, false);
                }
        }
    };

    // ================= 14 full chunks: no bounds checks anywhere ============
    for (int kc = 0; kc < K_FULL; kc += KC) {
        // A tile: 128 rows x 16 quads, 8 quads/thread, float4 loads
#pragma unroll
        for (int it = 0; it < 8; ++it) {
            int idx = tid + it * 256;
            int m = idx >> 4;
            int q = idx & 15;
            v4f v = *(const v4f*)(xb + (size_t)m * K_SZ + kc + q * 4);
            *(v4bf*)(&shA[m * KPAD + q * 4]) = __builtin_convertvector(v, v4bf);
        }
        // W tiles: 64 k x 16 quads, 4 quads/thread
#pragma unroll
        for (int it = 0; it < 4; ++it) {
            int idx = tid + it * 256;
            int qn = idx & 15;
            int k  = idx >> 4;
            stageW(kc + k, k, qn * 4);
        }
        __syncthreads();
        doWmma();
        __syncthreads();
    }

    // ================= tail chunk: kc = 896, 4 valid k ======================
    {
#pragma unroll
        for (int it = 0; it < 8; ++it) {
            int idx = tid + it * 256;
            int m = idx >> 4;
            int q = idx & 15;
            v4f v;
#pragma unroll
            for (int j = 0; j < 4; ++j) v[j] = 0.f;
            if (q == 0)
                v = *(const v4f*)(xb + (size_t)m * K_SZ + K_FULL);
            *(v4bf*)(&shA[m * KPAD + q * 4]) = __builtin_convertvector(v, v4bf);
        }
#pragma unroll
        for (int it = 0; it < 4; ++it) {
            int idx = tid + it * 256;
            int qn = idx & 15;
            int k  = idx >> 4;
            if (k < K_REM) {
                stageW(K_FULL + k, k, qn * 4);
            } else {
                bf16 z = (bf16)0.f;
#pragma unroll
                for (int j = 0; j < 4; ++j) {
                    shE[(qn * 4 + j) * KPAD + k] = z;
                    shG[(qn * 4 + j) * KPAD + k] = z;
                }
            }
        }
        __syncthreads();
        doWmma();
    }

    // ---- epilogue: biases (branch-free), bf16-round expert partial, add gen
    const int colLane = lane & 15;
    const int rowHalf = (lane & 16) ? 8 : 0;
#pragma unroll
    for (int ni = 0; ni < 2; ++ni) {
        const int col = n0 + wn + ni * 16 + colLane;
        float bE = 0.f;
#pragma unroll
        for (int e = 0; e < 8; ++e)
            bE += we[e] * bexp[e * D_SZ + col];
        const float bG = bgen[col] + bgen[D_SZ + col];
#pragma unroll
        for (int mi = 0; mi < 2; ++mi) {
            v8f aE = accE[mi * 2 + ni];
            v8f aG = accG[mi * 2 + ni];
#pragma unroll
            for (int i = 0; i < 8; ++i) {
                int m = wm + mi * 16 + rowHalf + i;
                out[((size_t)(b * L_SZ + m)) * D_SZ + col] =
                    bfround(aE[i] + bE) + aG[i] + bG;
            }
        }
    }
}

// ---------------------------------------------------------------------------
extern "C" void kernel_launch(void* const* d_in, const int* in_sizes, int n_in,
                              void* d_out, int out_size, void* d_ws, size_t ws_size,
                              hipStream_t stream) {
    (void)in_sizes; (void)n_in; (void)out_size; (void)ws_size;
    const float* x     = (const float*)d_in[0];
    const float* dkp   = (const float*)d_in[1];
    const float* Wexp  = (const float*)d_in[2];
    const float* bexp  = (const float*)d_in[3];
    const float* Wgen  = (const float*)d_in[4];
    const float* bgen  = (const float*)d_in[5];
    const float* Wgate = (const float*)d_in[6];
    const float* bgate = (const float*)d_in[7];
    float* out  = (float*)d_out;
    float* wBuf = (float*)d_ws;       // [512,8] gate weights

    moe_gate_kernel<<<dim3(B_SZ), dim3(256), 0, stream>>>(dkp, Wgate, bgate, wBuf);
    moe_mm_kernel<<<dim3(D_SZ / NT, B_SZ), dim3(256), 0, stream>>>(
        x, Wexp, bexp, Wgen, bgen, wBuf, out);
}
